// RNN_34488587387231
// MI455X (gfx1250) — matrix operand order repaired and checked by reading, not compiled, not run on hardware
//
#include <hip/hip_runtime.h>
#include <hip/hip_bf16.h>

typedef unsigned short u16;
typedef unsigned int   u32;
typedef __attribute__((ext_vector_type(16))) __bf16 v16bf;
typedef __attribute__((ext_vector_type(8)))  float  v8f;
typedef __attribute__((ext_vector_type(8)))  u32    v8u;

#define B_ 512
#define T_ 128
#define I_ 64
#define H_ 512
#define O_ 64
#define PL 300

// ---------- helpers ----------
__device__ __forceinline__ u16 f2bf(float f) {
  u32 u = __float_as_uint(f);
  u32 r = u + 0x7FFFu + ((u >> 16) & 1u);   // round-to-nearest-even
  return (u16)(r >> 16);
}
__device__ __forceinline__ v16bf ldmat(const u16* p) {
  return __builtin_bit_cast(v16bf, *(const v8u*)p);
}
__device__ __forceinline__ v8f wmma_bf16(v16bf a, v16bf b, v8f c) {
  // D = A(16x32 bf16) x B(32x16 bf16) + C(16x16 f32)
  return __builtin_amdgcn_wmma_f32_16x16x32_bf16(
      /*neg_a=*/false, a, /*neg_b=*/false, b,
      /*c_mod=*/(short)0, c, /*reuse_a=*/false, /*reuse_b=*/false);
}
__device__ __forceinline__ float sigmoidf_(float x) {
  return 1.0f / (1.0f + __expf(-x));
}
// CDNA5 async global->LDS copy (16B per lane), tracked by ASYNCcnt.
// vdst = LDS byte offset (LDS_BASE-relative), vaddr = 64-bit global address.
__device__ __forceinline__ void async_copy_b128(u32 ldsOff, const void* g) {
  asm volatile("global_load_async_to_lds_b128 %0, %1, off"
               :: "v"(ldsOff), "v"(g) : "memory");
}
__device__ __forceinline__ void wait_async0() {
  asm volatile("s_wait_asynccnt 0x0" ::: "memory");
}
__device__ __forceinline__ u32 lds_off(const void* shared_ptr) {
  // flat LDS address: addr[31:0] == wave-relative LDS byte offset (ISA 10.2)
  return (u32)(unsigned long long)shared_ptr;
}

// Packed tile conventions (wave32):
//  A tile (16 rows x 32 K, bf16): 512 u16, lane-major: lane L holds 16 u16 at L*16.
//  B tile (32 K x 16 N, bf16):  lane<16 -> N=lane, K=0..7,16..23 ; lane>=16 -> N=lane-16, K=8..15,24..31.
//  One 32B load per lane fills a v16bf WMMA operand.

// ---------- pack W^T into B-operand tiles: out[colTile][kTile][512] ----------
__global__ void pack_b_kernel(const float* __restrict__ W, u16* __restrict__ out,
                              int kTiles, int Kdim, int total) {
  int tid = blockIdx.x * blockDim.x + threadIdx.x;
  if (tid >= total) return;
  int tile = tid >> 9;
  int lane = (tid >> 4) & 31;
  int idx  = tid & 15;
  int c  = tile / kTiles;            // column tile over W rows (gate outputs)
  int k0 = tile % kTiles;            // K tile over W cols
  int N = c * 16 + (lane & 15);
  int K = k0 * 32 + ((idx < 8) ? ((lane >> 4) * 8 + idx) : (16 + (lane >> 4) * 8 + idx - 8));
  out[tid] = f2bf(W[N * Kdim + K]);  // B[K][N] = W[N][K]
}

// ---------- pack encoder inputs into A tiles: out[t][rowTile][kTile][512] ----------
__global__ void pack_x_kernel(const float* __restrict__ X, u16* __restrict__ out) {
  int tid = blockIdx.x * blockDim.x + threadIdx.x;   // T_*B_*I_ total
  int t    = tid >> 15;          // / (512*64)
  int rem  = tid & 32767;
  int tile = rem >> 9;           // 0..63 : rowTile*2 + kTile
  int lane = (rem >> 4) & 31;
  int idx  = rem & 15;
  int rt = tile >> 1, kt = tile & 1;
  int mm = lane & 15, hi = lane >> 4;
  int v = idx >> 1, h = idx & 1;
  int K = kt * 32 + ((v < 4) ? (v * 2 + h + hi * 8)
                             : (16 + (v - 4) * 2 + h + hi * 8));
  int row = rt * 16 + mm;
  out[tid] = f2bf(X[row * (T_ * I_) + t * I_ + K]);
}

// ---------- zero initial hidden state (fp32 + packed bf16) ----------
__global__ void zero_kernel(float* __restrict__ hF, u16* __restrict__ HA) {
  int tid = blockIdx.x * blockDim.x + threadIdx.x;   // 512*512 == 32*16*512
  hF[tid] = 0.0f;
  HA[tid] = 0;
}

// ---------- one fused GRU step: h' = GRU(h, x) ----------
// grid 256 x 128 threads; block = (rowGroup of 4 rowTiles) x (1 hidden col tile j)
// The 4 waves share j => gate B strips staged once per block into LDS via
// global_load_async_to_lds_b128 (ASYNCcnt), WMMA B operands then read from LDS.
__global__ __launch_bounds__(128)
void gru_step_kernel(const u16* __restrict__ XA,    // [32][2][512] packed A of x (or y)
                     const u16* __restrict__ HAin,  // [32][16][512] packed A of h
                     const float* __restrict__ hFin,// [512][512]
                     const u16* __restrict__ WhhB,  // [96][16][512]
                     const u16* __restrict__ WihB,  // [96][2][512]
                     const float* __restrict__ bih,
                     const float* __restrict__ bhh,
                     float* __restrict__ hFout,
                     u16* __restrict__ HAout) {
  __shared__ u16 smem[3 * 8192 + 3 * 1024];   // 54 KB: 3x16KB hidden B + 3x2KB input B
  const int tid  = threadIdx.x;
  const int lane = tid & 31;
  const int j    = blockIdx.x & 31;           // hidden-column tile (shared by block)
  const int rg   = blockIdx.x >> 5;           // row group 0..7
  const int rt   = rg * 4 + (tid >> 5);       // batch-row tile for this wave
  const int lo   = lane * 16;

  // ---- stage B operands into LDS (async, per-lane 16B chunks) ----
#pragma unroll
  for (int g = 0; g < 3; ++g) {               // hidden gates r,z,n: 16KB each
    const char* src = (const char*)(WhhB + (size_t)(g * 32 + j) * (16 * 512));
    u32 dst = lds_off(&smem[g * 8192]);
    for (int i = tid; i < 1024; i += 128)
      async_copy_b128(dst + i * 16, src + i * 16);
  }
  for (int i = tid; i < 384; i += 128) {      // input gates: 2KB each
    int g = i >> 7, c = i & 127;
    const char* src = (const char*)(WihB + (size_t)(g * 32 + j) * (2 * 512)) + c * 16;
    async_copy_b128(lds_off(&smem[24576 + g * 1024]) + c * 16, src);
  }
  wait_async0();
  __syncthreads();

  v8f ar = {}, az = {}, ani = {}, anh = {};

  // ---- hidden GEMM part: K = 512 (16 tiles); A from global, B from LDS ----
  const u16* ha  = HAin + (size_t)rt * (16 * 512);
  const u16* sbr = &smem[0];
  const u16* sbz = &smem[8192];
  const u16* sbn = &smem[16384];
#pragma unroll 4
  for (int k = 0; k < 16; ++k) {
    v16bf a  = ldmat(ha  + k * 512 + lo);
    v16bf wr = ldmat(sbr + k * 512 + lo);
    v16bf wz = ldmat(sbz + k * 512 + lo);
    v16bf wn = ldmat(sbn + k * 512 + lo);
    ar  = wmma_bf16(a, wr, ar);
    az  = wmma_bf16(a, wz, az);
    anh = wmma_bf16(a, wn, anh);
  }

  // ---- input GEMM part: K = 64 (2 tiles) ----
  const u16* xa   = XA + (size_t)rt * (2 * 512);
  const u16* sir  = &smem[24576];
  const u16* siz  = &smem[24576 + 1024];
  const u16* sinn = &smem[24576 + 2048];
#pragma unroll
  for (int k = 0; k < 2; ++k) {
    v16bf a = ldmat(xa + k * 512 + lo);
    ar  = wmma_bf16(a, ldmat(sir  + k * 512 + lo), ar);
    az  = wmma_bf16(a, ldmat(siz  + k * 512 + lo), az);
    ani = wmma_bf16(a, ldmat(sinn + k * 512 + lo), ani);
  }

  // ---- epilogue: gates + blend; write fp32 h' and re-pack bf16 A tile ----
  int colL = lane & 15, hi = lane >> 4;
  int col = j * 16 + colL;                    // hidden column 0..511
  float bR  = bih[col] + bhh[col];
  float bZ  = bih[H_ + col] + bhh[H_ + col];
  float biN = bih[2 * H_ + col];
  float bhN = bhh[2 * H_ + col];

  int cc = col & 31, kt = col >> 5;
  int laneHi, v;
  if      (cc < 8)  { laneHi = 0; v = cc >> 1; }
  else if (cc < 16) { laneHi = 1; v = (cc - 8) >> 1; }
  else if (cc < 24) { laneHi = 0; v = 4 + ((cc - 16) >> 1); }
  else              { laneHi = 1; v = 4 + ((cc - 24) >> 1); }
  int hh = cc & 1;
  u16* haOut = HAout + (size_t)(rt * 16 + kt) * 512;

#pragma unroll
  for (int r = 0; r < 8; ++r) {
    int row = rt * 16 + r + 8 * hi;
    float g_r = sigmoidf_(ar[r] + bR);
    float g_z = sigmoidf_(az[r] + bZ);
    float n   = tanhf(ani[r] + biN + g_r * (anh[r] + bhN));
    float hold = hFin[row * H_ + col];
    float hnew = (1.0f - g_z) * n + g_z * hold;
    hFout[row * H_ + col] = hnew;
    haOut[(laneHi * 16 + r + 8 * hi) * 16 + v * 2 + hh] = f2bf(hnew);
  }
}

// ---------- output projection: y = h @ Wo^T + bo ; emit to d_out and packed YA ----------
// grid 32 x 128 = 128 waves; wave = rowTile(32) x oTile(4)
__global__ __launch_bounds__(128)
void proj_kernel(const u16* __restrict__ HA,   // [32][16][512]
                 const u16* __restrict__ WoB,  // [4][16][512]
                 const float* __restrict__ bo,
                 float* __restrict__ out,      // [B][PL][O]
                 u16* __restrict__ YA,         // [32][2][512]
                 int tIdx) {
  int wave = blockIdx.x * (blockDim.x >> 5) + (threadIdx.x >> 5);
  int lane = threadIdx.x & 31;
  int rt = wave >> 2;
  int o  = wave & 3;
  int lo = lane * 16;

  v8f acc = {};
  const u16* ha = HA  + (size_t)rt * (16 * 512);
  const u16* wb = WoB + (size_t)o  * (16 * 512);
#pragma unroll 4
  for (int k = 0; k < 16; ++k) {
    v16bf a = ldmat(ha + k * 512 + lo);
    v16bf b = ldmat(wb + k * 512 + lo);
    acc = wmma_bf16(a, b, acc);
  }

  int colL = lane & 15, hi = lane >> 4;
  int col = o * 16 + colL;                  // 0..63
  float bias = bo[col];
  int cc = col & 31, kt = col >> 5;
  int laneHi, v;
  if      (cc < 8)  { laneHi = 0; v = cc >> 1; }
  else if (cc < 16) { laneHi = 1; v = (cc - 8) >> 1; }
  else if (cc < 24) { laneHi = 0; v = 4 + ((cc - 16) >> 1); }
  else              { laneHi = 1; v = 4 + ((cc - 24) >> 1); }
  int hh = cc & 1;
  u16* yaOut = YA + (size_t)(rt * 2 + kt) * 512;

#pragma unroll
  for (int r = 0; r < 8; ++r) {
    int row = rt * 16 + r + 8 * hi;
    float y = acc[r] + bias;
    out[(row * PL + tIdx) * O_ + col] = y;
    yaOut[(laneHi * 16 + r + 8 * hi) * 16 + v * 2 + hh] = f2bf(y);
  }
}

// ---------- host orchestration ----------
extern "C" void kernel_launch(void* const* d_in, const int* in_sizes, int n_in,
                              void* d_out, int out_size, void* d_ws, size_t ws_size,
                              hipStream_t stream) {
  (void)in_sizes; (void)n_in; (void)out_size; (void)ws_size;
  const float* input = (const float*)d_in[0];   // [B,T,I]
  const float* Wih   = (const float*)d_in[1];   // [3H,I]
  const float* Whh   = (const float*)d_in[2];   // [3H,H]
  const float* bih   = (const float*)d_in[3];   // [3H]
  const float* bhh   = (const float*)d_in[4];   // [3H]
  const float* Wo    = (const float*)d_in[5];   // [O,H]
  const float* bo    = (const float*)d_in[6];   // [O]
  float* out = (float*)d_out;                   // [B,PL,O]

  char* base = (char*)d_ws;
  size_t off = 0;
  auto alloc = [&](size_t bytes) -> void* {
    void* r = base + off;
    off = (off + bytes + 1023) & ~(size_t)1023;
    return r;
  };
  u16*   WhhB  = (u16*)alloc((size_t)96 * 16 * 512 * 2);
  u16*   WihB  = (u16*)alloc((size_t)96 * 2  * 512 * 2);
  u16*   WoB   = (u16*)alloc((size_t)4  * 16 * 512 * 2);
  u16*   XA    = (u16*)alloc((size_t)T_ * 32 * 2 * 512 * 2);
  float* hF0   = (float*)alloc((size_t)B_ * H_ * 4);
  float* hF1   = (float*)alloc((size_t)B_ * H_ * 4);
  u16*   HA0   = (u16*)alloc((size_t)32 * 16 * 512 * 2);
  u16*   HA1   = (u16*)alloc((size_t)32 * 16 * 512 * 2);
  u16*   YA    = (u16*)alloc((size_t)32 * 2 * 512 * 2);
  float* hF[2] = { hF0, hF1 };
  u16*   HA[2] = { HA0, HA1 };

  // one-time packs (inside graph; cheap vs 427 steps)
  pack_b_kernel<<<(96 * 16 * 512) / 256, 256, 0, stream>>>(Whh, WhhB, 16, H_, 96 * 16 * 512);
  pack_b_kernel<<<(96 * 2  * 512) / 256, 256, 0, stream>>>(Wih, WihB, 2,  I_, 96 * 2 * 512);
  pack_b_kernel<<<(4  * 16 * 512) / 256, 256, 0, stream>>>(Wo,  WoB,  16, H_, 4 * 16 * 512);
  pack_x_kernel<<<(T_ * B_ * I_) / 256, 256, 0, stream>>>(input, XA);
  zero_kernel<<<(B_ * H_) / 256, 256, 0, stream>>>(hF[0], HA[0]);

  int cur = 0;
  // encoder: 128 steps
  for (int t = 0; t < T_; ++t) {
    gru_step_kernel<<<256, 128, 0, stream>>>(XA + (size_t)t * 32 * 2 * 512,
                                             HA[cur], hF[cur], WhhB, WihB, bih, bhh,
                                             hF[cur ^ 1], HA[cur ^ 1]);
    cur ^= 1;
  }
  // y0 = proj(hT)
  proj_kernel<<<32, 128, 0, stream>>>(HA[cur], WoB, bo, out, YA, 0);
  // autoregressive decode
  for (int s = 1; s < PL; ++s) {
    gru_step_kernel<<<256, 128, 0, stream>>>(YA, HA[cur], hF[cur],
                                             WhhB, WihB, bih, bhh,
                                             hF[cur ^ 1], HA[cur ^ 1]);
    cur ^= 1;
    proj_kernel<<<32, 128, 0, stream>>>(HA[cur], WoB, bo, out, YA, s);
  }
}
